// MultiHeadAttentionByPspp_81157702025386
// MI455X (gfx1250) — matrix operand order repaired and checked by reading, not compiled, not run on hardware
//
#include <hip/hip_runtime.h>
#include <hip/hip_bf16.h>
#include <stdint.h>

// Problem constants (match reference): B=2, T=2048, DM=1024, H=16, HD=64
#define B_  2
#define T_  2048
#define DM_ 1024
#define H_  16
#define HD_ 64

typedef __attribute__((ext_vector_type(16))) __bf16 v16bf;
typedef __attribute__((ext_vector_type(8)))  float  v8f;

union FragBF {
  v16bf v;
  uint4 q[2];
  unsigned short s[16];
};

__device__ __forceinline__ unsigned short f2bf(float f) {
  union { float f; unsigned u; } x; x.f = f;
  unsigned u = x.u;
  unsigned r = (u + 0x7FFFu + ((u >> 16) & 1u)) >> 16;   // round-to-nearest-even
  return (unsigned short)r;
}

__device__ __forceinline__ v8f wmma_bf16(const FragBF& a, const FragBF& b, v8f c) {
  // D = A(16x32 bf16) * B(32x16 bf16) + C(16x16 f32)
  return __builtin_amdgcn_wmma_f32_16x16x32_bf16(false, a.v, false, b.v, (short)0, c, false, false);
}

// LDS weight-tile geometry shared by the two GEMM kernels:
// 64 rows (N) x 32 cols (K) bf16, row stride padded to 40 elements (80B) to
// spread the 64 LDS banks across the 16-lane fragment reads.
#define BT_STRIDE 40
#define BT_ELEMS  (64 * BT_STRIDE)

// ---------------------------------------------------------------- K0: f32 -> bf16
__global__ void cvt_bf16_kernel(const float* __restrict__ in,
                                unsigned short* __restrict__ out, int n) {
  int i = blockIdx.x * blockDim.x + threadIdx.x;
  int stride = gridDim.x * blockDim.x;
  for (; i < n; i += stride) out[i] = f2bf(in[i]);
}

// ---------------------------------------------------------------- K1: QKV GEMM + RoPE
// qkv[b,t,e] = sum_d x[b,t,d] * w_qkv[e,d]; e = s*1024 + h*64 + d
// Block = 8 waves sharing one 64-wide e-slot; weight tile double-buffered in
// LDS via global_load_async_to_lds_b128 (ASYNCcnt). Each wave: 16(t) x 64(e).
// Outputs: Qb,Kb = bf16 [B*H][T][64];  Vt = bf16 [B*H][64][T]
__global__ __launch_bounds__(256) void qkv_rope_kernel(
    const unsigned short* __restrict__ xb,     // [B*T][1024] bf16
    const unsigned short* __restrict__ wqkvb,  // [3072][1024] bf16
    const float* __restrict__ cosp,            // [T][64]
    const float* __restrict__ sinp,            // [T][64]
    unsigned short* __restrict__ Qb,
    unsigned short* __restrict__ Kb,
    unsigned short* __restrict__ Vt) {
  __shared__ __align__(16) unsigned short btile[2][BT_ELEMS];

  const int tid  = threadIdx.x;
  const int lane = tid & 31;
  const int wave = tid >> 5;
  const int eslot  = blockIdx.x >> 5;          // 0..47 = s*16 + h
  const int mgroup = blockIdx.x & 31;          // 32 groups of 8 row-tiles
  const int s  = eslot >> 4;
  const int h  = eslot & 15;
  const int e0 = s * 1024 + h * 64;
  const int m0 = (mgroup * 8 + wave) * 16;
  const int half = lane >> 4;
  const int lan  = lane & 15;

  // async-copy mapping: 256 threads x 16B = 4KB tile (64 rows x 32 bf16)
  const int crow = tid >> 2;                   // 0..63
  const int cchk = tid & 3;                    // 4 x 8-element chunks
  const unsigned short* gsrc = wqkvb + (size_t)(e0 + crow) * 1024 + cchk * 8;
  unsigned ldst[2];
  ldst[0] = (unsigned)(size_t)(const void*)&btile[0][0]
          + (unsigned)(crow * (BT_STRIDE * 2) + cchk * 16);
  ldst[1] = ldst[0] + (unsigned)(BT_ELEMS * 2);

  auto issueB = [&](int k, int buf) {
    unsigned long long ga = (unsigned long long)(size_t)(gsrc + k);
    asm volatile("global_load_async_to_lds_b128 %0, %1, off"
                 :: "v"(ldst[buf]), "v"(ga)
                 : "memory");
  };

  v8f acc[4] = {};
  const unsigned short* arow = xb + (size_t)(m0 + lan) * 1024;

  issueB(0, 0);
  for (int ks = 0; ks < 32; ks++) {
    const int k = ks * 32;
    asm volatile("s_wait_asynccnt 0x0" ::: "memory");  // my tile copy landed
    __syncthreads();                                   // everyone's landed; prior reads done
    if (ks + 1 < 32) issueB(k + 32, (ks + 1) & 1);     // overlap next copy with compute

    FragBF a;
    const uint4* pa = (const uint4*)(arow + k + 8 * half);
    a.q[0] = pa[0];            // j=0..7  -> K = j + 8*half
    a.q[1] = pa[2];            // j=8..15 -> K = j + 8 + 8*half

    const unsigned short* bt = &btile[ks & 1][0];
    for (int n = 0; n < 4; n++) {
      FragBF b;
      const uint4* pb = (const uint4*)(bt + (n * 16 + lan) * BT_STRIDE + 8 * half);
      b.q[0] = pb[0];          // j=0..7 -> K = j + 8*half; j=8..15 -> K = j + 8 + 8*half
      b.q[1] = pb[2];
      acc[n] = wmma_bf16(a, b, acc[n]);
    }
  }

  const int bb    = m0 / T_;
  const int tbase = m0 % T_;
  const size_t bh = (size_t)bb * H_ + h;

  if (s == 2) {                // V -> transposed [bh][d][t]
    for (int n = 0; n < 4; n++) {
      int d = n * 16 + lan;
      for (int r = 0; r < 8; r++) {
        int t = tbase + r + 8 * half;
        Vt[(bh * 64 + d) * T_ + t] = f2bf(acc[n][r]);
      }
    }
  } else {                     // Q or K with RoPE
    unsigned short* dst = (s == 0) ? Qb : Kb;
    for (int n = 0; n < 4; n++) {
      int d = n * 16 + lan;
      for (int r = 0; r < 8; r++) {
        int t = tbase + r + 8 * half;
        float c  = cosp[t * 64 + d];
        float sn = sinp[t * 64 + d];
        float u  = acc[n][r];
        float up = acc[(n + 2) & 3][r];        // partner dim d +/- 32
        float rot = (n < 2) ? -up : up;        // rotate_half sign
        dst[(bh * T_ + t) * 64 + d] = f2bf(u * c + rot * sn);
      }
    }
  }
}

// ---------------------------------------------------------------- K2: causal flash attention
// One wave per (b,h, 16-query tile). 32 keys per iteration.
__global__ __launch_bounds__(256) void attn_kernel(
    const unsigned short* __restrict__ Qb,   // [B*H][T][64]
    const unsigned short* __restrict__ Kb,   // [B*H][T][64]
    const unsigned short* __restrict__ Vt,   // [B*H][64][T]
    unsigned short* __restrict__ Ob) {       // [B*T][1024] bf16
  __shared__ __align__(16) unsigned short plds[8][16 * 32];  // per-wave P tile

  const int lane = threadIdx.x & 31;
  const int wave = threadIdx.x >> 5;
  const int gid  = blockIdx.x * 8 + wave;    // 4096 waves total
  const int bh   = gid >> 7;                 // / (T/16)
  const int qt   = gid & 127;
  const int q0   = qt * 16;
  const int half = lane >> 4;
  const int lan  = lane & 15;

  // Q A-fragments (two 32-wide d chunks)
  FragBF qf[2];
  const unsigned short* qrow = Qb + ((size_t)bh * T_ + q0 + lan) * 64;
  for (int c = 0; c < 2; c++) {
    const uint4* p = (const uint4*)(qrow + c * 32 + 8 * half);
    qf[c].q[0] = p[0];
    qf[c].q[1] = p[2];
  }

  float mrow[8], lrow[8];
  for (int r = 0; r < 8; r++) { mrow[r] = -3.0e38f; lrow[r] = 0.0f; }
  v8f o[4] = {};

  const int nkb = (q0 + 16 + 31) >> 5;       // causal key-block count
  for (int kb = 0; kb < nkb; kb++) {
    const int k0 = kb * 32;

    if (kb + 1 < nkb)
      __builtin_prefetch(Kb + ((size_t)bh * T_ + k0 + 32 + lan) * 64, 0, 0);

    // S = Q * K^T (scaled) for two 16-key tiles
    v8f sacc[2] = {};
    for (int kt = 0; kt < 2; kt++) {
      const unsigned short* krow = Kb + ((size_t)bh * T_ + k0 + kt * 16 + lan) * 64;
      for (int c = 0; c < 2; c++) {
        FragBF kf;
        const uint4* p = (const uint4*)(krow + c * 32 + 8 * half);
        kf.q[0] = p[0];
        kf.q[1] = p[2];
        sacc[kt] = wmma_bf16(qf[c], kf, sacc[kt]);
      }
    }

    // scale + causal mask + row max
    float rmax[8];
    for (int r = 0; r < 8; r++) {
      const int qidx = q0 + r + 8 * half;
      float s0 = sacc[0][r] * 0.125f;
      float s1 = sacc[1][r] * 0.125f;
      if (k0 + lan      > qidx) s0 = -3.0e38f;
      if (k0 + 16 + lan > qidx) s1 = -3.0e38f;
      sacc[0][r] = s0; sacc[1][r] = s1;
      float v = fmaxf(s0, s1);
      for (int off = 1; off < 16; off <<= 1) v = fmaxf(v, __shfl_xor(v, off, 16));
      rmax[r] = v;
    }

    // online softmax update; P written to LDS in bf16
    for (int r = 0; r < 8; r++) {
      float mnew  = fmaxf(mrow[r], rmax[r]);
      float alpha = __expf(mrow[r] - mnew);
      mrow[r] = mnew;
      float p0 = __expf(sacc[0][r] - mnew);
      float p1 = __expf(sacc[1][r] - mnew);
      float rs = p0 + p1;
      for (int off = 1; off < 16; off <<= 1) rs += __shfl_xor(rs, off, 16);
      lrow[r] = lrow[r] * alpha + rs;
      for (int n = 0; n < 4; n++) o[n][r] *= alpha;
      const int row = r + 8 * half;
      plds[wave][row * 32 + lan]      = f2bf(p0);
      plds[wave][row * 32 + 16 + lan] = f2bf(p1);
    }

    // P A-fragment from LDS (DS ops are in-order within a wave)
    FragBF pf;
    {
      const uint4* p = (const uint4*)(&plds[wave][lan * 32 + 8 * half]);
      pf.q[0] = p[0];
      pf.q[1] = p[2];
    }

    // O += P * V  (V transposed -> contiguous B-fragment loads)
    for (int n = 0; n < 4; n++) {
      FragBF vf;
      const uint4* p = (const uint4*)(Vt + ((size_t)bh * 64 + n * 16 + lan) * T_ + k0 + 8 * half);
      vf.q[0] = p[0];
      vf.q[1] = p[2];
      o[n] = wmma_bf16(pf, vf, o[n]);
    }
  }

  // normalize and store O as bf16 [b][t][h*64+d]
  const size_t b = bh >> 4;
  const int    h = bh & 15;
  for (int n = 0; n < 4; n++) {
    for (int r = 0; r < 8; r++) {
      int t = q0 + r + 8 * half;
      float val = o[n][r] / lrow[r];
      Ob[(b * T_ + t) * DM_ + h * 64 + n * 16 + lan] = f2bf(val);
    }
  }
}

// ---------------------------------------------------------------- K3: out projection
// out[m,e] = sum_d Ob[m,d] * w_proj[e,d]; block = 8 waves sharing one 64-wide
// e-slot, weight tile double-buffered in LDS via async load-to-LDS.
__global__ __launch_bounds__(256) void proj_kernel(
    const unsigned short* __restrict__ Ob,   // [B*T][1024] bf16
    const unsigned short* __restrict__ wb,   // [1024][1024] bf16
    float* __restrict__ out) {               // [B*T][1024] f32
  __shared__ __align__(16) unsigned short btile[2][BT_ELEMS];

  const int tid  = threadIdx.x;
  const int lane = tid & 31;
  const int wave = tid >> 5;
  const int nslot  = blockIdx.x >> 5;        // 0..15 (64 cols each)
  const int mgroup = blockIdx.x & 31;
  const int e0 = nslot * 64;
  const int m0 = (mgroup * 8 + wave) * 16;
  const int half = lane >> 4;
  const int lan  = lane & 15;

  const int crow = tid >> 2;
  const int cchk = tid & 3;
  const unsigned short* gsrc = wb + (size_t)(e0 + crow) * 1024 + cchk * 8;
  unsigned ldst[2];
  ldst[0] = (unsigned)(size_t)(const void*)&btile[0][0]
          + (unsigned)(crow * (BT_STRIDE * 2) + cchk * 16);
  ldst[1] = ldst[0] + (unsigned)(BT_ELEMS * 2);

  auto issueB = [&](int k, int buf) {
    unsigned long long ga = (unsigned long long)(size_t)(gsrc + k);
    asm volatile("global_load_async_to_lds_b128 %0, %1, off"
                 :: "v"(ldst[buf]), "v"(ga)
                 : "memory");
  };

  v8f acc[4] = {};
  const unsigned short* arow = Ob + (size_t)(m0 + lan) * 1024;

  issueB(0, 0);
  for (int ks = 0; ks < 32; ks++) {
    const int k = ks * 32;
    asm volatile("s_wait_asynccnt 0x0" ::: "memory");
    __syncthreads();
    if (ks + 1 < 32) issueB(k + 32, (ks + 1) & 1);

    FragBF a;
    const uint4* pa = (const uint4*)(arow + k + 8 * half);
    a.q[0] = pa[0];
    a.q[1] = pa[2];

    const unsigned short* bt = &btile[ks & 1][0];
    for (int n = 0; n < 4; n++) {
      FragBF b;
      const uint4* pb = (const uint4*)(bt + (n * 16 + lan) * BT_STRIDE + 8 * half);
      b.q[0] = pb[0];
      b.q[1] = pb[2];
      acc[n] = wmma_bf16(a, b, acc[n]);
    }
  }

  for (int n = 0; n < 4; n++)
    for (int r = 0; r < 8; r++)
      out[(size_t)(m0 + r + 8 * half) * 1024 + e0 + n * 16 + lan] = acc[n][r];
}

// ---------------------------------------------------------------- launch
extern "C" void kernel_launch(void* const* d_in, const int* in_sizes, int n_in,
                              void* d_out, int out_size, void* d_ws, size_t ws_size,
                              hipStream_t stream) {
  const float* x      = (const float*)d_in[0];   // [2,2048,1024]
  const float* cosp   = (const float*)d_in[1];   // [2048,64]
  const float* sinp   = (const float*)d_in[2];   // [2048,64]
  const float* w_qkv  = (const float*)d_in[3];   // [3072,1024]
  const float* w_proj = (const float*)d_in[4];   // [1024,1024]
  float* out          = (float*)d_out;           // [2,2048,1024]

  char* ws = (char*)d_ws;
  const size_t n_x     = (size_t)B_ * T_ * DM_;         // 4,194,304
  const size_t n_wqkv  = (size_t)3 * DM_ * DM_;         // 3,145,728
  const size_t n_wproj = (size_t)DM_ * DM_;             // 1,048,576
  const size_t n_head  = (size_t)B_ * H_ * T_ * HD_;    // 4,194,304

  unsigned short* xb     = (unsigned short*)(ws);
  unsigned short* wqkvb  = (unsigned short*)(ws + 2 * n_x);
  unsigned short* wprojb = (unsigned short*)(ws + 2 * (n_x + n_wqkv));
  unsigned short* Qb     = (unsigned short*)(ws + 2 * (n_x + n_wqkv + n_wproj));
  unsigned short* Kb     = (unsigned short*)(ws + 2 * (n_x + n_wqkv + n_wproj + n_head));
  unsigned short* Vt     = (unsigned short*)(ws + 2 * (n_x + n_wqkv + n_wproj + 2 * n_head));
  unsigned short* Ob     = (unsigned short*)(ws + 2 * (n_x + n_wqkv + n_wproj + 3 * n_head));

  cvt_bf16_kernel<<<1024, 256, 0, stream>>>(x, xb, (int)n_x);
  cvt_bf16_kernel<<<1024, 256, 0, stream>>>(w_qkv, wqkvb, (int)n_wqkv);
  cvt_bf16_kernel<<<1024, 256, 0, stream>>>(w_proj, wprojb, (int)n_wproj);

  // 48 e-slots * 32 m-groups = 1536 blocks of 8 waves
  qkv_rope_kernel<<<1536, 256, 0, stream>>>(xb, wqkvb, cosp, sinp, Qb, Kb, Vt);

  // B*H * (T/16) = 4096 waves -> 512 blocks
  attn_kernel<<<512, 256, 0, stream>>>(Qb, Kb, Vt, Ob);

  // 16 n-slots * 32 m-groups = 512 blocks
  proj_kernel<<<512, 256, 0, stream>>>(Ob, wprojb, out);
}
